// MultiHeadAttention_13365938225152
// MI455X (gfx1250) — matrix operand order repaired and checked by reading, compiled, not run on hardware
//
#include <hip/hip_runtime.h>
#include <hip/hip_bf16.h>

// ---------------------------------------------------------------------------
// MI455X (gfx1250) multi-head attention:
//   bf16 WMMA (v_wmma_f32_16x16x32_bf16) for all matmuls, f32 accumulation,
//   f32 online softmax, flash-attention style streaming over keys.
// ---------------------------------------------------------------------------

typedef __attribute__((ext_vector_type(16))) __bf16 bf16x16;
typedef __attribute__((ext_vector_type(8)))  __bf16 bf16x8;
typedef __attribute__((ext_vector_type(8)))  float  v8f;

#define D_MODEL 1024
#define S_LEN   2048
#define BATCH   2
#define NHEADS  16
#define DK      64
#define M_ROWS  (BATCH * S_LEN)   // 4096

union AFrag { bf16x16 v; bf16x8 h[2]; };

__device__ __forceinline__ v8f wmma_bf16(bf16x16 a, bf16x16 b, v8f c) {
    // D = A(16x32 bf16) x B(32x16 bf16) + C(16x16 f32)
    return __builtin_amdgcn_wmma_f32_16x16x32_bf16(
        /*neg_a=*/false, a, /*neg_b=*/false, b,
        /*c_mod=*/(short)0, c, /*reuse_a=*/false, /*reuse_b=*/false);
}

enum { MODE_QK = 0, MODE_VT = 1, MODE_OUT = 2 };

// ---------------------------------------------------------------------------
// GEMM: Y = X[4096x1024](f32) @ W[1024x1024](f32) + bias
//   MODE_QK : store bf16 as [B,H,S,64]
//   MODE_VT : store bf16 as [B,H,64,S]   (transposed, for PV B-operand)
//   MODE_OUT: store f32  as [4096,1024]
// Block: 256 threads (8 waves), macro tile 128x128, K-step 32.
// Waves: wm = wave&3 -> 32-row chunk, wn = wave>>2 -> 64-col chunk.
// ---------------------------------------------------------------------------
template <int MODE>
__global__ __launch_bounds__(256) void gemm_kernel(
    const float* __restrict__ X, const float* __restrict__ W,
    const float* __restrict__ bias, void* __restrict__ out)
{
    __shared__ __align__(16) __bf16 As[128][32];   // [m][k] 8KB
    __shared__ __align__(16) __bf16 Bs[128][32];   // [n][k] 8KB (transposed)

    const int t    = threadIdx.x;
    const int lane = t & 31;
    const int wave = t >> 5;
    const int wm   = wave & 3;    // row chunk 0..3  (32 rows each)
    const int wn   = wave >> 2;   // col chunk 0..1  (64 cols each)
    const int m0   = blockIdx.y * 128;
    const int n0   = blockIdx.x * 128;

    const int l16 = lane & 15;
    const int lh  = lane >> 4;     // 0/1
    const int kb8  = lh * 16;      // A-fragment K base (contiguous k = 16*lh + s)
    const int kb16 = lh * 16;      // B-fragment K base (contiguous, same as A)

    v8f acc[2][4];
#pragma unroll
    for (int mt = 0; mt < 2; ++mt)
#pragma unroll
        for (int nt = 0; nt < 4; ++nt)
#pragma unroll
            for (int r = 0; r < 8; ++r) acc[mt][nt][r] = 0.0f;

    for (int k0 = 0; k0 < D_MODEL; k0 += 32) {
        // ---- stage A tile: 128x32 f32 -> bf16 LDS --------------------------
#pragma unroll
        for (int i = 0; i < 4; ++i) {
            const int linear = t + i * 256;          // float4 index, 0..1023
            const int row = linear >> 3;             // 8 float4 per row
            const int c4  = (linear & 7) * 4;
            const float4 f = *(const float4*)(X + (size_t)(m0 + row) * D_MODEL + k0 + c4);
            As[row][c4 + 0] = (__bf16)f.x;
            As[row][c4 + 1] = (__bf16)f.y;
            As[row][c4 + 2] = (__bf16)f.z;
            As[row][c4 + 3] = (__bf16)f.w;
        }
        // ---- stage B tile: 32x128 f32 -> bf16 LDS, transposed to [n][k] ----
#pragma unroll
        for (int i = 0; i < 4; ++i) {
            const int linear = t + i * 256;          // float4 index, 0..1023
            const int kr = linear >> 5;              // 32 float4 per k-row
            const int c4 = (linear & 31) * 4;
            const float4 f = *(const float4*)(W + (size_t)(k0 + kr) * D_MODEL + n0 + c4);
            Bs[c4 + 0][kr] = (__bf16)f.x;
            Bs[c4 + 1][kr] = (__bf16)f.y;
            Bs[c4 + 2][kr] = (__bf16)f.z;
            Bs[c4 + 3][kr] = (__bf16)f.w;
        }
        __syncthreads();

        // ---- fragments (contiguous k = 16*lh + s for A and B) -------------------
        AFrag a[2];
#pragma unroll
        for (int mt = 0; mt < 2; ++mt) {
            const __bf16* p = &As[wm * 32 + mt * 16 + l16][kb8];
            a[mt].h[0] = *(const bf16x8*)(p);        // K = kb8 .. kb8+7
            a[mt].h[1] = *(const bf16x8*)(p + 8);    // K = kb8+8 .. kb8+15
        }
        bf16x16 bf[4];
#pragma unroll
        for (int nt = 0; nt < 4; ++nt)
            bf[nt] = *(const bf16x16*)(&Bs[wn * 64 + nt * 16 + l16][kb16]);

#pragma unroll
        for (int mt = 0; mt < 2; ++mt)
#pragma unroll
            for (int nt = 0; nt < 4; ++nt)
                acc[mt][nt] = wmma_bf16(a[mt].v, bf[nt], acc[mt][nt]);

        __syncthreads();
    }

    // ---- epilogue: bias + store per mode -----------------------------------
#pragma unroll
    for (int mt = 0; mt < 2; ++mt) {
#pragma unroll
        for (int nt = 0; nt < 4; ++nt) {
            const int gn = n0 + wn * 64 + nt * 16 + l16;
            const float bv = bias[gn];
            const int gmb = m0 + wm * 32 + mt * 16 + lh * 8;   // rows gmb..gmb+7
            if (MODE == MODE_OUT) {
                float* o = (float*)out;
#pragma unroll
                for (int r = 0; r < 8; ++r)
                    o[(size_t)(gmb + r) * D_MODEL + gn] = acc[mt][nt][r] + bv;
            } else if (MODE == MODE_QK) {
                __bf16* o = (__bf16*)out;
#pragma unroll
                for (int r = 0; r < 8; ++r) {
                    const int gm = gmb + r;
                    const int b = gm >> 11, s = gm & (S_LEN - 1);
                    const int h = gn >> 6,  d = gn & (DK - 1);
                    o[(((size_t)(b * NHEADS + h) * S_LEN + s) << 6) + d] = (__bf16)(acc[mt][nt][r] + bv);
                }
            } else { // MODE_VT: [B,H,64,S]; 8 consecutive s -> one 16B store
                const int b = gmb >> 11, s = gmb & (S_LEN - 1);
                const int h = gn >> 6,   d = gn & (DK - 1);
                bf16x8 pk;
#pragma unroll
                for (int r = 0; r < 8; ++r) pk[r] = (__bf16)(acc[mt][nt][r] + bv);
                __bf16* o = (__bf16*)out;
                *(bf16x8*)(&o[((size_t)(b * NHEADS + h) * DK + d) * S_LEN + s]) = pk;
            }
        }
    }
}

// ---------------------------------------------------------------------------
// Flash attention: one 16-query tile per wave, 8 waves/block, 64-key blocks.
//   Q,K : bf16 [B,H,S,64]   (A / B operands load straight from global in
//                            WMMA lane layout: contiguous 16B / 32B per lane)
//   Vt  : bf16 [B,H,64,S]   (B operand for PV, contiguous per lane)
//   out : f32 [4096,1024]   ([b*S+s][h*64+d], heads re-interleaved)
// ---------------------------------------------------------------------------
__global__ __launch_bounds__(256) void attn_kernel(
    const __bf16* __restrict__ Q, const __bf16* __restrict__ K,
    const __bf16* __restrict__ Vt, const int* __restrict__ mask,
    float* __restrict__ outws)
{
    __shared__ __align__(16) __bf16 Plds[8][16][64];   // per-wave P tile, 16KB

    const int t    = threadIdx.x;
    const int lane = t & 31;
    const int wave = t >> 5;
    const int l16  = lane & 15;
    const int lh   = lane >> 4;

    const int bh = blockIdx.y;               // 0..31
    const int b  = bh >> 4;
    const int h  = bh & (NHEADS - 1);
    const int q0 = blockIdx.x * 128 + wave * 16;

    const __bf16* Qb = Q  + (size_t)bh * S_LEN * DK;
    const __bf16* Kb = K  + (size_t)bh * S_LEN * DK;
    const __bf16* Vb = Vt + (size_t)bh * DK * S_LEN;
    const int* mrow  = mask + (size_t)b * S_LEN * S_LEN;

    // Q fragments (A-operand): lane row = l16, contiguous k = 16*lh + s per 32-chunk
    AFrag qf[2];
    {
        const __bf16* qr = Qb + (size_t)(q0 + l16) * DK;
#pragma unroll
        for (int kk = 0; kk < 2; ++kk) {
            qf[kk].h[0] = *(const bf16x8*)(qr + kk * 32 + lh * 16);
            qf[kk].h[1] = *(const bf16x8*)(qr + kk * 32 + lh * 16 + 8);
        }
    }

    float rowmax[8], rowsum[8];
    v8f accO[4];
#pragma unroll
    for (int r = 0; r < 8; ++r) { rowmax[r] = -3.0e38f; rowsum[r] = 0.0f; }
#pragma unroll
    for (int nt = 0; nt < 4; ++nt)
#pragma unroll
        for (int r = 0; r < 8; ++r) accO[nt][r] = 0.0f;

    for (int kb = 0; kb < S_LEN; kb += 64) {
        // ---- scores: S = (Q Kt) * 1/8, masked ------------------------------
        v8f sa[4];
#pragma unroll
        for (int nt = 0; nt < 4; ++nt) {
            const int j0 = kb + nt * 16;
            const __bf16* kr = Kb + (size_t)(j0 + l16) * DK;  // B lane: n = l16
            const bf16x16 k0v = *(const bf16x16*)(kr + lh * 16);        // feats 0..31
            const bf16x16 k1v = *(const bf16x16*)(kr + 32 + lh * 16);   // feats 32..63
            v8f s = {0.f, 0.f, 0.f, 0.f, 0.f, 0.f, 0.f, 0.f};
            s = wmma_bf16(qf[0].v, k0v, s);
            s = wmma_bf16(qf[1].v, k1v, s);
#pragma unroll
            for (int r = 0; r < 8; ++r) {
                const int qm = q0 + lh * 8 + r;
                const int mv = mrow[(size_t)qm * S_LEN + j0 + l16];
                s[r] = (mv == 0) ? -1.0e12f : s[r] * 0.125f;
            }
            sa[nt] = s;
        }

        // ---- online softmax (rows live across 16-lane halves) --------------
        float newmax[8], corr[8];
#pragma unroll
        for (int r = 0; r < 8; ++r) {
            float m = fmaxf(fmaxf(sa[0][r], sa[1][r]), fmaxf(sa[2][r], sa[3][r]));
#pragma unroll
            for (int off = 1; off < 16; off <<= 1)
                m = fmaxf(m, __shfl_xor(m, off, 32));
            newmax[r] = fmaxf(rowmax[r], m);
            corr[r]   = __expf(rowmax[r] - newmax[r]);
        }
        float bsum[8];
#pragma unroll
        for (int r = 0; r < 8; ++r) bsum[r] = 0.0f;
#pragma unroll
        for (int nt = 0; nt < 4; ++nt)
#pragma unroll
            for (int r = 0; r < 8; ++r) {
                const float p = __expf(sa[nt][r] - newmax[r]);
                bsum[r] += p;
                Plds[wave][lh * 8 + r][nt * 16 + l16] = (__bf16)p;
            }
#pragma unroll
        for (int r = 0; r < 8; ++r) {
            float s = bsum[r];
#pragma unroll
            for (int off = 1; off < 16; off <<= 1)
                s += __shfl_xor(s, off, 32);
            rowsum[r] = rowsum[r] * corr[r] + s;
            rowmax[r] = newmax[r];
        }
#pragma unroll
        for (int nt = 0; nt < 4; ++nt)
#pragma unroll
            for (int r = 0; r < 8; ++r) accO[nt][r] *= corr[r];

        __syncthreads();   // P(C-layout f32) -> LDS -> P(A-layout bf16)

        AFrag pf[2];
        {
            const __bf16* pr = &Plds[wave][l16][0];
#pragma unroll
            for (int kk = 0; kk < 2; ++kk) {
                pf[kk].h[0] = *(const bf16x8*)(pr + kk * 32 + lh * 16);
                pf[kk].h[1] = *(const bf16x8*)(pr + kk * 32 + lh * 16 + 8);
            }
        }

        // ---- O += P @ V  (V transposed: lane reads 16 contiguous keys) -----
#pragma unroll
        for (int nt = 0; nt < 4; ++nt) {
            const __bf16* vr = Vb + (size_t)(nt * 16 + l16) * S_LEN + kb;
            const bf16x16 v0 = *(const bf16x16*)(vr + lh * 16);        // keys kb+0..31
            const bf16x16 v1 = *(const bf16x16*)(vr + 32 + lh * 16);   // keys kb+32..63
            accO[nt] = wmma_bf16(pf[0].v, v0, accO[nt]);
            accO[nt] = wmma_bf16(pf[1].v, v1, accO[nt]);
        }
        __syncthreads();   // before next block overwrites Plds
    }

    // ---- normalize + store [b*S+s][h*64+d] f32 -----------------------------
#pragma unroll
    for (int nt = 0; nt < 4; ++nt) {
        const int col = h * DK + nt * 16 + l16;
#pragma unroll
        for (int r = 0; r < 8; ++r) {
            const int qm = q0 + lh * 8 + r;
            outws[(size_t)(b * S_LEN + qm) * D_MODEL + col] = accO[nt][r] / rowsum[r];
        }
    }
}

// ---------------------------------------------------------------------------
extern "C" void kernel_launch(void* const* d_in, const int* in_sizes, int n_in,
                              void* d_out, int out_size, void* d_ws, size_t ws_size,
                              hipStream_t stream) {
    (void)in_sizes; (void)n_in; (void)out_size; (void)ws_size;

    const float* query = (const float*)d_in[0];
    const float* key   = (const float*)d_in[1];
    const float* value = (const float*)d_in[2];
    const int*   mask  = (const int*)  d_in[3];
    const float* Wq    = (const float*)d_in[4];
    const float* bq    = (const float*)d_in[5];
    const float* Wk    = (const float*)d_in[6];
    const float* bk    = (const float*)d_in[7];
    const float* Wv    = (const float*)d_in[8];
    const float* bv    = (const float*)d_in[9];
    const float* Wo    = (const float*)d_in[10];
    const float* bo    = (const float*)d_in[11];

    const size_t QKV_ELEMS = (size_t)BATCH * NHEADS * S_LEN * DK;  // 4,194,304
    char*   ws  = (char*)d_ws;
    __bf16* qws = (__bf16*)ws;
    __bf16* kws = qws + QKV_ELEMS;
    __bf16* vws = kws + QKV_ELEMS;
    float*  aws = (float*)(ws + 3 * QKV_ELEMS * sizeof(__bf16));   // 25,165,824 B offset

    const dim3 gemm_grid(D_MODEL / 128, M_ROWS / 128);   // (8, 32)
    const dim3 attn_grid(S_LEN / 128, BATCH * NHEADS);   // (16, 32)
    const dim3 block(256);

    gemm_kernel<MODE_QK><<<gemm_grid, block, 0, stream>>>(query, Wq, bq, (void*)qws);
    gemm_kernel<MODE_QK><<<gemm_grid, block, 0, stream>>>(key,   Wk, bk, (void*)kws);
    gemm_kernel<MODE_VT><<<gemm_grid, block, 0, stream>>>(value, Wv, bv, (void*)vws);
    attn_kernel<<<attn_grid, block, 0, stream>>>(qws, kws, vws, mask, aws);
    gemm_kernel<MODE_OUT><<<gemm_grid, block, 0, stream>>>(aws, Wo, bo, d_out);
}
